// GATMC_34651796144597
// MI455X (gfx1250) — matrix operand order repaired and checked by reading, not compiled, not run on hardware
//
#include <hip/hip_runtime.h>
#include <hip/hip_bf16.h>
#include <stdint.h>

// ---------------------------------------------------------------------------
// GAT 3-layer pipeline for MI455X (gfx1250, wave32, WMMA).
//  - Activations/weights pre-cast to bf16; GEMM = v_wmma_f32_16x16x32_bf16,
//    128x128 block tile, 8 waves (4Mx2N), wave tile 32x64 (8 WMMA / K-step).
//  - LDS double-buffered staging via GLOBAL_LOAD_ASYNC_TO_LDS_B128 (ASYNCcnt),
//    overlapping DMA of tile k+1 with WMMA on tile k.
//  - Edge softmax + scatter: block-per-edge, f32 global atomics.
// ---------------------------------------------------------------------------

typedef __attribute__((ext_vector_type(16))) __bf16          bf16x16;
typedef __attribute__((ext_vector_type(8)))  float           f32x8;
typedef __attribute__((ext_vector_type(4)))  float           f32x4;
typedef __attribute__((ext_vector_type(4)))  unsigned short  u16x4;
typedef __attribute__((ext_vector_type(8)))  unsigned short  u16x8;
typedef __attribute__((ext_vector_type(16))) unsigned short  u16x16;
typedef __attribute__((ext_vector_type(4)))  int             i32x4;

#define GAT_N 50000
#define GAT_E 1600000
#define GAT_H 4

__device__ __forceinline__ unsigned short f2bf(float f) {
    union { float f; unsigned u; } v; v.f = f;
    unsigned u = v.u;
    u += 0x7fffu + ((u >> 16) & 1u);            // round-to-nearest-even
    return (unsigned short)(u >> 16);
}

__device__ __forceinline__ void atomicMaxF32(float* addr, float v) {
    if (v >= 0.0f) atomicMax((int*)addr, __float_as_int(v));
    else           atomicMin((unsigned int*)addr, __float_as_uint(v));
}

#if defined(__gfx1250__) && __has_builtin(__builtin_amdgcn_global_load_async_to_lds_b128)
#define GAT_ASYNC_LDS 1
typedef __attribute__((address_space(1))) i32x4 as1_i32x4;
typedef __attribute__((address_space(3))) i32x4 as3_i32x4;
#else
#define GAT_ASYNC_LDS 0
#endif

// copy 16B global->LDS (per lane)
__device__ __forceinline__ void stage16(const unsigned short* g, unsigned short* l) {
#if GAT_ASYNC_LDS
    __builtin_amdgcn_global_load_async_to_lds_b128(
        (as1_i32x4*)g, (as3_i32x4*)l, 0, 0);
#else
    *(u16x8*)l = *(const u16x8*)g;
#endif
}

// ---------------------------------------------------------------------------
// bf16 WMMA GEMM: C[M,Nout] = A[M,K] * Bt[Nout,K]^T  (A row-major bf16,
// Bt = B transposed, row-major bf16, C fp32 row-major).
// Block: 256 threads = 8 waves. Block tile 128(M) x 128(N), K step 32,
// LDS double-buffered async staging.
// ---------------------------------------------------------------------------
__global__ __launch_bounds__(256) void gemm_bf16_wmma(
    const unsigned short* __restrict__ A, const unsigned short* __restrict__ Bt,
    float* __restrict__ C, int M, int K, int Nout)
{
    __shared__ __align__(16) unsigned short As[2][128 * 48];   // [m][k] stride 48
    __shared__ __align__(16) unsigned short Bs[2][128 * 48];   // [n][k] stride 48

    const int tid  = threadIdx.x;
    const int lane = tid & 31;
    const int wid  = tid >> 5;
    const int wm   = wid & 3;      // 4 M-waves (32 rows each)
    const int wn   = wid >> 2;     // 2 N-waves (64 cols each)
    const int bm   = blockIdx.y * 128;
    const int bn   = blockIdx.x * 128;

    f32x8 acc[2][4] = {};

    // stage A (128x32) + B (128x32) tiles for K-offset k0 into buffer `buf`
    auto issue_tile = [&](int k0, int buf) {
        #pragma unroll
        for (int p = 0; p < 2; ++p) {
            int q   = tid + p * 256;        // 0..511
            int row = q >> 2;               // 0..127
            int c8  = (q & 3) * 8;          // k chunk
            int ga  = bm + row; if (ga >= M) ga = M - 1;   // clamp; masked at writeout
            stage16(A  + (size_t)ga * K + k0 + c8,         &As[buf][row * 48 + c8]);
            stage16(Bt + (size_t)(bn + row) * K + k0 + c8, &Bs[buf][row * 48 + c8]);
        }
    };

    const int nk = K >> 5;
    issue_tile(0, 0);

    for (int kt = 0; kt < nk; ++kt) {
        const int cur = kt & 1;
        const bool has_next = (kt + 1) < nk;
        if (has_next) issue_tile((kt + 1) << 5, cur ^ 1);
#if GAT_ASYNC_LDS
        // 4 async ops/wave per tile: allow next tile's 4 to remain in flight.
        if (has_next) asm volatile("s_wait_asynccnt 4" ::: "memory");
        else          asm volatile("s_wait_asynccnt 0" ::: "memory");
#endif
        __syncthreads();

        // A 16x32 bf16 frag: lane l holds row m=l%16; lanes<16: K {0..7,16..23},
        // lanes>=16: K {8..15,24..31}.
        auto ldA = [&](int ms) -> bf16x16 {
            int row  = wm * 32 + ms * 16 + (lane & 15);
            int koff = (lane >> 4) * 8;
            const unsigned short* p = &As[cur][row * 48 + koff];
            u16x8 lo = *(const u16x8*)p;
            u16x8 hi = *(const u16x8*)(p + 16);
            u16x16 u;
            #pragma unroll
            for (int j = 0; j < 8; ++j) { u[j] = lo[j]; u[8 + j] = hi[j]; }
            return __builtin_bit_cast(bf16x16, u);
        };
        // B 32x16 bf16 frag: lane l holds col n=l%16; lanes<16: K {0..7,16..23},
        // lanes>=16: K {8..15,24..31}.
        auto ldB = [&](int ns) -> bf16x16 {
            int col = wn * 64 + ns * 16 + (lane & 15);
            int ko  = (lane >> 4) * 8;
            const unsigned short* p = &Bs[cur][col * 48 + ko];
            u16x8 lo = *(const u16x8*)p;
            u16x8 hi = *(const u16x8*)(p + 16);
            u16x16 u;
            #pragma unroll
            for (int j = 0; j < 8; ++j) { u[j] = lo[j]; u[8 + j] = hi[j]; }
            return __builtin_bit_cast(bf16x16, u);
        };

        bf16x16 a0 = ldA(0), a1 = ldA(1);
        #pragma unroll
        for (int ns = 0; ns < 4; ++ns) {
            bf16x16 b = ldB(ns);
            acc[0][ns] = __builtin_amdgcn_wmma_f32_16x16x32_bf16(false, a0, false, b, (short)0, acc[0][ns], false, false);
            acc[1][ns] = __builtin_amdgcn_wmma_f32_16x16x32_bf16(false, a1, false, b, (short)0, acc[1][ns], false, false);
        }
        __syncthreads();   // protect buf `cur` before it is re-staged at kt+1
    }

    // C/D layout: VGPR r -> M=r (lanes 0-15) / M=8+r (lanes 16-31), N = lane%16.
    const int rofs = (lane >> 4) << 3;
    const int col0 = bn + wn * 64 + (lane & 15);
    const bool interior = (bm + 128 <= M);
    #pragma unroll
    for (int ms = 0; ms < 2; ++ms) {
        #pragma unroll
        for (int ns = 0; ns < 4; ++ns) {
            int row0 = bm + wm * 32 + ms * 16 + rofs;
            float* p = C + (size_t)row0 * Nout + col0 + ns * 16;
            if (interior) {
                #pragma unroll
                for (int r = 0; r < 8; ++r) { *p = acc[ms][ns][r]; p += Nout; }
            } else {
                #pragma unroll
                for (int r = 0; r < 8; ++r) {
                    if (row0 + r < M) *p = acc[ms][ns][r];
                    p += Nout;
                }
            }
        }
    }
}

// ---------------------------------------------------------------------------
// fp32 -> bf16 casts
// ---------------------------------------------------------------------------
__global__ __launch_bounds__(256) void cast_bf16_k(
    const float* __restrict__ in, unsigned short* __restrict__ out, size_t n4)
{
    size_t i = (size_t)blockIdx.x * blockDim.x + threadIdx.x;
    if (i >= n4) return;
    f32x4 v = ((const f32x4*)in)[i];
    u16x4 o = { f2bf(v.x), f2bf(v.y), f2bf(v.z), f2bf(v.w) };
    ((u16x4*)out)[i] = o;
}

// B [K,Nout] f32 -> Bt [Nout,K] bf16 (coalesced reads, scattered 2B writes; tiny)
__global__ __launch_bounds__(256) void cast_t_bf16_k(
    const float* __restrict__ B, unsigned short* __restrict__ Bt, int K, int Nout)
{
    int i = blockIdx.x * blockDim.x + threadIdx.x;
    if (i >= K * Nout) return;
    int k = i / Nout, n = i % Nout;
    Bt[(size_t)n * K + k] = f2bf(B[i]);
}

// ---------------------------------------------------------------------------
// el[n,h] = dot(feat[n,h,:], al[h,:]),  er likewise. One wave32 per (n,h).
// ---------------------------------------------------------------------------
__global__ __launch_bounds__(256) void attn_dots(
    const float* __restrict__ feat, const float* __restrict__ al,
    const float* __restrict__ ar, float* __restrict__ el, float* __restrict__ er,
    int Nn, int D)
{
    int gw   = (blockIdx.x * blockDim.x + threadIdx.x) >> 5;
    int lane = threadIdx.x & 31;
    if (gw >= Nn * GAT_H) return;
    int n = gw / GAT_H, h = gw % GAT_H;
    const float* f   = feat + (size_t)n * GAT_H * D + h * D;
    const float* pal = al + h * D;
    const float* par = ar + h * D;
    float sl = 0.f, sr = 0.f;
    for (int d = lane; d < D; d += 32) {
        float x = f[d];
        sl += x * pal[d];
        sr += x * par[d];
    }
    #pragma unroll
    for (int off = 16; off > 0; off >>= 1) {
        sl += __shfl_xor(sl, off, 32);
        sr += __shfl_xor(sr, off, 32);
    }
    if (lane == 0) { el[gw] = sl; er[gw] = sr; }
}

// ---------------------------------------------------------------------------
// Edge softmax pieces (vectorized over the 4 heads)
// ---------------------------------------------------------------------------
__global__ __launch_bounds__(256) void edge_max_k(
    const int* __restrict__ src, const int* __restrict__ dst,
    const float* __restrict__ el, const float* __restrict__ er,
    float* __restrict__ emax, int E)
{
    int e = blockIdx.x * blockDim.x + threadIdx.x;
    if (e >= E) return;
    int s = src[e], d = dst[e];
    f32x4 vl = *(const f32x4*)(el + (size_t)s * GAT_H);
    f32x4 vr = *(const f32x4*)(er + (size_t)d * GAT_H);
    #pragma unroll
    for (int h = 0; h < GAT_H; ++h) {
        float v = vl[h] + vr[h];
        v = (v >= 0.f) ? v : 0.2f * v;
        atomicMaxF32(&emax[d * GAT_H + h], v);
    }
}

__global__ __launch_bounds__(256) void edge_expsum_k(
    const int* __restrict__ src, const int* __restrict__ dst,
    const float* __restrict__ el, const float* __restrict__ er,
    const float* __restrict__ emax, float* __restrict__ ee,
    float* __restrict__ denom, int E)
{
    int e = blockIdx.x * blockDim.x + threadIdx.x;
    if (e >= E) return;
    int s = src[e], d = dst[e];
    f32x4 vl = *(const f32x4*)(el + (size_t)s * GAT_H);
    f32x4 vr = *(const f32x4*)(er + (size_t)d * GAT_H);
    f32x4 vm = *(const f32x4*)(emax + (size_t)d * GAT_H);
    f32x4 o;
    #pragma unroll
    for (int h = 0; h < GAT_H; ++h) {
        float v = vl[h] + vr[h];
        v = (v >= 0.f) ? v : 0.2f * v;
        o[h] = __expf(v - vm[h]);
        atomicAdd(&denom[d * GAT_H + h], o[h]);
    }
    *(f32x4*)(ee + (size_t)e * GAT_H) = o;
}

// alpha[e,h] = ee[e,h] / max(denom[dst[e],h], 1e-9)   (in-place on ee)
__global__ __launch_bounds__(256) void edge_alpha_k(
    const int* __restrict__ dst, const float* __restrict__ denom,
    float* __restrict__ ee, int E)
{
    int e = blockIdx.x * blockDim.x + threadIdx.x;
    if (e >= E) return;
    int d = dst[e];
    f32x4 dn = *(const f32x4*)(denom + (size_t)d * GAT_H);
    f32x4 v  = *(const f32x4*)(ee + (size_t)e * GAT_H);
    #pragma unroll
    for (int h = 0; h < GAT_H; ++h) v[h] = v[h] / fmaxf(dn[h], 1e-9f);
    *(f32x4*)(ee + (size_t)e * GAT_H) = v;
}

// msg: one block per edge (blockDim = H*D/4); src/dst become scalar loads.
__global__ void edge_msg_k(
    const int* __restrict__ src, const int* __restrict__ dst,
    const float* __restrict__ feat, const float* __restrict__ alpha,
    float* __restrict__ acc, int dshift)
{
    int e  = blockIdx.x;
    int s  = src[e], d = dst[e];
    int r4 = threadIdx.x * 4;
    int h  = r4 >> dshift;
    float a = alpha[(size_t)e * GAT_H + h];
    int F  = blockDim.x * 4;
    f32x4 f = *(const f32x4*)(feat + (size_t)s * F + r4);
    float* o = acc + (size_t)d * F + r4;
    atomicAdd(o + 0, f.x * a);
    atomicAdd(o + 1, f.y * a);
    atomicAdd(o + 2, f.z * a);
    atomicAdd(o + 3, f.w * a);
}

// ---------------------------------------------------------------------------
// Epilogues + init
// ---------------------------------------------------------------------------
__global__ __launch_bounds__(256) void fill_k(float* __restrict__ p, float v, size_t n) {
    size_t i = (size_t)blockIdx.x * blockDim.x + threadIdx.x;
    if (i < n) p[i] = v;
}

__global__ __launch_bounds__(256) void elu_k(
    const float* __restrict__ acc, const float* __restrict__ res,
    float* __restrict__ out, size_t n)
{
    size_t i = (size_t)blockIdx.x * blockDim.x + threadIdx.x;
    if (i >= n) return;
    float v = acc[i] + (res ? res[i] : 0.f);
    out[i] = (v > 0.f) ? v : (__expf(v) - 1.f);
}

__global__ __launch_bounds__(256) void final_k(
    const float* __restrict__ acc, const float* __restrict__ res,
    float* __restrict__ out, int Nn, int D)
{
    int i = blockIdx.x * blockDim.x + threadIdx.x;
    if (i >= Nn * D) return;
    int n = i / D, d = i % D;
    float s = 0.f;
    #pragma unroll
    for (int h = 0; h < GAT_H; ++h) {
        size_t idx = (size_t)n * GAT_H * D + h * D + d;
        s += tanhf(acc[idx] + res[idx]);
    }
    out[i] = tanhf(s * 0.25f);
}

// ---------------------------------------------------------------------------
// Orchestration
// ---------------------------------------------------------------------------
static inline int cdiv(long long a, long long b) { return (int)((a + b - 1) / b); }

extern "C" void kernel_launch(void* const* d_in, const int* in_sizes, int n_in,
                              void* d_out, int out_size, void* d_ws, size_t ws_size,
                              hipStream_t stream)
{
    const float* x     = (const float*)d_in[0];
    const int*   src   = (const int*)d_in[1];
    const int*   dst   = (const int*)d_in[2];
    const float* W0    = (const float*)d_in[3];
    const float* al0   = (const float*)d_in[4];
    const float* ar0   = (const float*)d_in[5];
    const float* W1    = (const float*)d_in[6];
    const float* al1   = (const float*)d_in[7];
    const float* ar1   = (const float*)d_in[8];
    const float* W2    = (const float*)d_in[9];
    const float* al2   = (const float*)d_in[10];
    const float* ar2   = (const float*)d_in[11];
    const float* Wres2 = (const float*)d_in[12];

    const int N = GAT_N, E = GAT_E;
    const size_t NF = (size_t)N * 1024;

    float* bufA  = (float*)d_ws;          // [N,1024]
    float* bufB  = bufA + NF;             // [N,1024]
    float* bufC  = bufB + NF;             // [N,1024]
    float* ee    = bufC + NF;             // [E,4]
    float* el    = ee + (size_t)E * GAT_H;
    float* er    = el + (size_t)N * GAT_H;
    float* emax  = er + (size_t)N * GAT_H;
    float* denom = emax + (size_t)N * GAT_H;
    unsigned short* Abf  = (unsigned short*)(denom + (size_t)N * GAT_H); // [N,1024] bf16
    unsigned short* Btbf = Abf + NF;                                     // [1024,1024] bf16
    (void)ws_size; (void)in_sizes; (void)n_in; (void)out_size;

    const float NEG_INF = -__builtin_inff();
    const int nh = N * GAT_H;
    const int nhB = cdiv(nh, 256);
    const int eB  = cdiv(E, 256);
    const dim3 blk(256);

    auto cast_act = [&](const float* p, size_t n) {
        cast_bf16_k<<<cdiv((long long)(n / 4), 256), blk, 0, stream>>>(p, Abf, n / 4);
    };
    auto cast_w = [&](const float* w, int K, int Nout) {
        cast_t_bf16_k<<<cdiv((long long)K * Nout, 256), blk, 0, stream>>>(w, Btbf, K, Nout);
    };
    auto run_gemm = [&](float* Cdst, int K, int Nout) {
        gemm_bf16_wmma<<<dim3(Nout / 128, cdiv(N, 128)), blk, 0, stream>>>(
            Abf, Btbf, Cdst, N, K, Nout);
    };

    auto softmax_and_msg = [&](const float* feat, float* acc, int D, int dshift) {
        fill_k<<<nhB, blk, 0, stream>>>(emax, NEG_INF, (size_t)nh);
        fill_k<<<nhB, blk, 0, stream>>>(denom, 0.f, (size_t)nh);
        edge_max_k<<<eB, blk, 0, stream>>>(src, dst, el, er, emax, E);
        edge_expsum_k<<<eB, blk, 0, stream>>>(src, dst, el, er, emax, ee, denom, E);
        edge_alpha_k<<<eB, blk, 0, stream>>>(dst, denom, ee, E);
        edge_msg_k<<<E, dim3(GAT_H * D / 4), 0, stream>>>(src, dst, feat, ee, acc, dshift);
    };

    // ---------------- Layer 0: feat0 = x @ W0 ; h1 = elu(agg) ----------------
    cast_act(x, (size_t)N * 128);
    cast_w(W0, 128, 1024);
    run_gemm(bufA, 128, 1024);
    attn_dots<<<cdiv((long long)nh * 32, 256), blk, 0, stream>>>(bufA, al0, ar0, el, er, N, 256);
    fill_k<<<cdiv(NF, 256), blk, 0, stream>>>(bufB, 0.f, NF);
    softmax_and_msg(bufA, bufB, 256, 8);                 // acc -> bufB
    elu_k<<<cdiv(NF, 256), blk, 0, stream>>>(bufB, nullptr, bufB, NF);   // h1 in bufB

    // ---------------- Layer 1: feat1 = h1 @ W1 ; h2 = elu(agg + h1) ----------
    cast_act(bufB, NF);
    cast_w(W1, 1024, 1024);
    run_gemm(bufA, 1024, 1024);
    attn_dots<<<cdiv((long long)nh * 32, 256), blk, 0, stream>>>(bufA, al1, ar1, el, er, N, 256);
    fill_k<<<cdiv(NF, 256), blk, 0, stream>>>(bufC, 0.f, NF);
    softmax_and_msg(bufA, bufC, 256, 8);                 // acc -> bufC
    elu_k<<<cdiv(NF, 256), blk, 0, stream>>>(bufC, bufB, bufA, NF);      // h2 in bufA

    // ---------------- Layer 2: feat2 = h2 @ W2 ; res2 = h2 @ Wres2 ----------
    const size_t NF2 = (size_t)N * 512;
    float* feat2 = bufB;
    float* res2  = bufC;
    float* acc2  = bufB + NF2;
    cast_act(bufA, NF);
    cast_w(W2, 1024, 512);
    run_gemm(feat2, 1024, 512);
    cast_w(Wres2, 1024, 512);
    run_gemm(res2, 1024, 512);
    attn_dots<<<cdiv((long long)nh * 32, 256), blk, 0, stream>>>(feat2, al2, ar2, el, er, N, 128);
    fill_k<<<cdiv(NF2, 256), blk, 0, stream>>>(acc2, 0.f, NF2);
    softmax_and_msg(feat2, acc2, 128, 7);                // acc -> acc2
    final_k<<<cdiv((long long)N * 128, 256), blk, 0, stream>>>(acc2, res2, (float*)d_out, N, 128);
}
